// QuantizedBasicBlock_60017872995059
// MI455X (gfx1250) — 8-bit matrix operand order repaired and checked by reading, compiled, not run on hardware
//
#include <hip/hip_runtime.h>


typedef __attribute__((ext_vector_type(2))) int v2i;
typedef __attribute__((ext_vector_type(4))) int v4i;
typedef __attribute__((ext_vector_type(8))) int v8i;

#define BB   64
#define CC   64
#define HH   56
#define WW   56
#define HWX  (HH * WW)
#define NPIX (BB * HWX)

__device__ __forceinline__ float clamp8(float v) {
    return fminf(fmaxf(v, -128.f), 127.f);
}

__global__ void pack_x_kernel(const float* __restrict__ x,
                              const int* __restrict__ zxp,
                              signed char* __restrict__ out8) {
    int tid = blockIdx.x * blockDim.x + threadIdx.x;
    if (tid >= NPIX * 16) return;
    int c4  = tid / NPIX;
    int pix = tid - c4 * NPIX;
    int b   = pix / HWX;
    int hw  = pix - b * HWX;
    int zx  = *zxp;
    const float* px = x + (size_t)b * CC * HWX + hw;
    int d = 0;
#pragma unroll
    for (int j = 0; j < 4; ++j) {
        int c = c4 * 4 + j;
        int v = (int)px[(size_t)c * HWX] - zx;
        d |= (int)((unsigned char)(signed char)v) << (8 * j);
    }
    ((int*)out8)[(size_t)pix * 16 + c4] = d;
}

__global__ void pack_w_kernel(const float* __restrict__ w,
                              int* __restrict__ frag) {
    int tid = blockIdx.x * blockDim.x + threadIdx.x;
    if (tid >= 9 * 4 * 32 * 8) return;
    int v    = tid & 7;
    int lane = (tid >> 3) & 31;
    int mt   = (tid >> 8) & 3;
    int t    = tid >> 10;
    int kh = t / 3, kw = t % 3;
    int half = lane >> 4;
    int co   = mt * 16 + (lane & 15);
    int K    = ((v >> 2) << 5) + half * 16 + ((v & 3) << 2);
    int d = 0;
#pragma unroll
    for (int j = 0; j < 4; ++j) {
        int ci  = K + j;
        int val = (int)w[(((size_t)co * CC + ci) * 3 + kh) * 3 + kw];
        d |= (int)((unsigned char)(signed char)val) << (8 * j);
    }
    frag[tid] = d;
}

__global__ void prep_scales_kernel(const int* __restrict__ M0b1,
                                   const int* __restrict__ shb1,
                                   const int* __restrict__ M0b2,
                                   const int* __restrict__ shb2,
                                   float* __restrict__ sbn1,
                                   float* __restrict__ sbn2) {
    int c = threadIdx.x;
    if (c < CC) {
        sbn1[c] = ldexpf((float)M0b1[c], -31 - shb1[c]);
        sbn2[c] = ldexpf((float)M0b2[c], -31 - shb2[c]);
    }
}

template <int FINAL>
__global__ void __launch_bounds__(256)
conv_wmma_kernel(const signed char* __restrict__ in8,
                 const int*  __restrict__ wfrag,
                 const float* __restrict__ bias,
                 const int*  __restrict__ M0c, const int* __restrict__ shc,
                 const float* __restrict__ g,  const float* __restrict__ be,
                 const float* __restrict__ sbn,
                 const int*  __restrict__ zmp,
                 signed char* __restrict__ mid8,
                 const signed char* __restrict__ x8,
                 const int* __restrict__ M0a1, const int* __restrict__ sha1,
                 const int* __restrict__ M0a2, const int* __restrict__ sha2,
                 const int* __restrict__ zaddp,
                 float* __restrict__ outf) {
    const int lane = threadIdx.x & 31;
    const int wav  = threadIdx.x >> 5;
    const int mt   = wav & 3;
    const int grp  = wav >> 2;
    const int half = lane >> 4;
    const int col  = lane & 15;
    const int pbase = blockIdx.x * 128 + grp * 64;

    int pix[4], ohv[4], owv[4], obase[4];
#pragma unroll
    for (int nt = 0; nt < 4; ++nt) {
        int p  = pbase + nt * 16 + col;
        int b_ = p / HWX;
        int hw = p - b_ * HWX;
        int oh = hw / WW;
        pix[nt]   = p;
        ohv[nt]   = oh;
        owv[nt]   = hw - oh * WW;
        obase[nt] = b_ * CC * HWX + hw;
    }

    __builtin_prefetch(in8 + (size_t)pix[0] * CC, 0, 0);

    v8i acc[4];
#pragma unroll
    for (int nt = 0; nt < 4; ++nt) acc[nt] = (v8i){0, 0, 0, 0, 0, 0, 0, 0};

    const int* afp = wfrag + (mt * 32 + lane) * 8;

#pragma unroll
    for (int t = 0; t < 9; ++t) {
        const int kh = t / 3, kw = t % 3;

        const int* ap = afp + t * 1024;
        v4i alo = *(const v4i*)ap;
        v4i ahi = *(const v4i*)(ap + 4);
        v8i afrag = __builtin_shufflevector(alo, ahi, 0, 1, 2, 3, 4, 5, 6, 7);

#pragma unroll
        for (int nt = 0; nt < 4; ++nt) {
            const int ih = ohv[nt] + kh - 1;
            const int iw = owv[nt] + kw - 1;

            v8i bfrag = {0, 0, 0, 0, 0, 0, 0, 0};
            if ((unsigned)ih < (unsigned)HH && (unsigned)iw < (unsigned)WW) {
                const signed char* p =
                    in8 + (size_t)(pix[nt] + (kh - 1) * WW + (kw - 1)) * CC +
                    half * 16;
                v4i blo = *(const v4i*)p;
                v4i bhi = *(const v4i*)(p + 32);
                bfrag = __builtin_shufflevector(blo, bhi, 0, 1, 2, 3, 4, 5, 6, 7);
            }

            acc[nt] = __builtin_amdgcn_wmma_i32_16x16x64_iu8(
                true, afrag, true, bfrag, acc[nt], false, false);
        }
    }

    const int   cobase = mt * 16 + half * 8;
    const float zm = (float)*zmp;
    const float sc = ldexpf((float)*M0c, -31 - *shc);

    if (FINAL == 0) {
#pragma unroll
        for (int nt = 0; nt < 4; ++nt) {
            int d0 = 0, d1 = 0;
#pragma unroll
            for (int r = 0; r < 8; ++r) {
                int co   = cobase + r;
                float t1 = clamp8(rintf(((float)acc[nt][r] + bias[co]) * sc) + zm);
                float t2 = clamp8(rintf(((t1 - zm) * g[co] + be[co]) * sbn[co]) + zm);
                int   q  = (int)clamp8(t2 - zm);
                int   u  = (int)(unsigned char)(signed char)q;
                if (r < 4) d0 |= u << (8 * r); else d1 |= u << (8 * (r - 4));
            }
            v2i dv = {d0, d1};
            *(v2i*)(mid8 + (size_t)pix[nt] * CC + cobase) = dv;
        }
    } else {
        const float sa1 = ldexpf((float)*M0a1, -31 - *sha1);
        const float sa2 = ldexpf((float)*M0a2, -31 - *sha2);
        const float za  = (float)*zaddp;
#pragma unroll
        for (int nt = 0; nt < 4; ++nt) {
            const signed char* idp = x8 + (size_t)pix[nt] * CC + cobase;
            float* op = outf + obase[nt];
#pragma unroll
            for (int r = 0; r < 8; ++r) {
                int co   = cobase + r;
                float t1 = clamp8(rintf(((float)acc[nt][r] + bias[co]) * sc) + zm);
                float t2 = clamp8(rintf(((t1 - zm) * g[co] + be[co]) * sbn[co]) + zm);
                float outr = rintf((t2 - zm) * sa2);
                float idr  = rintf((float)idp[r] * sa1);
                op[(size_t)co * HWX] =
                    fminf(fmaxf(idr + outr + za, -128.f), 127.f);
            }
        }
    }
}

extern "C" void kernel_launch(void* const* d_in, const int* in_sizes, int n_in,
                              void* d_out, int out_size, void* d_ws, size_t ws_size,
                              hipStream_t stream) {
    const float* x    = (const float*)d_in[0];
    const float* w1   = (const float*)d_in[1];
    const float* b1   = (const float*)d_in[2];
    const float* w2   = (const float*)d_in[3];
    const float* b2   = (const float*)d_in[4];
    const float* g1   = (const float*)d_in[5];
    const float* be1  = (const float*)d_in[6];
    const float* g2   = (const float*)d_in[7];
    const float* be2  = (const float*)d_in[8];
    const int*   z_x  = (const int*)d_in[9];
    const int*   z_mid= (const int*)d_in[10];
    const int*   M0c1 = (const int*)d_in[11];
    const int*   shc1 = (const int*)d_in[12];
    const int*   M0b1 = (const int*)d_in[13];
    const int*   shb1 = (const int*)d_in[14];
    const int*   M0c2 = (const int*)d_in[15];
    const int*   shc2 = (const int*)d_in[16];
    const int*   M0b2 = (const int*)d_in[17];
    const int*   shb2 = (const int*)d_in[18];
    const int*   M0a1 = (const int*)d_in[19];
    const int*   sha1 = (const int*)d_in[20];
    const int*   M0a2 = (const int*)d_in[21];
    const int*   sha2 = (const int*)d_in[22];
    const int*   zadd = (const int*)d_in[23];

    const size_t ACT = (size_t)NPIX * CC;
    char* ws = (char*)d_ws;
    signed char* in8x = (signed char*)ws;
    signed char* mid8 = (signed char*)(ws + ACT);
    int*   wf1  = (int*)(ws + 2 * ACT);
    int*   wf2  = (int*)(ws + 2 * ACT + 36864);
    float* sbn1 = (float*)(ws + 2 * ACT + 2 * 36864);
    float* sbn2 = (float*)(ws + 2 * ACT + 2 * 36864 + 256);

    pack_x_kernel<<<(NPIX * 16) / 256, 256, 0, stream>>>(x, z_x, in8x);
    pack_w_kernel<<<36, 256, 0, stream>>>(w1, wf1);
    pack_w_kernel<<<36, 256, 0, stream>>>(w2, wf2);
    prep_scales_kernel<<<1, 64, 0, stream>>>(M0b1, shb1, M0b2, shb2, sbn1, sbn2);

    conv_wmma_kernel<0><<<NPIX / 128, 256, 0, stream>>>(
        in8x, wf1, b1, M0c1, shc1, g1, be1, sbn1, z_mid,
        mid8, nullptr, nullptr, nullptr, nullptr, nullptr, nullptr, nullptr);

    conv_wmma_kernel<1><<<NPIX / 128, 256, 0, stream>>>(
        mid8, wf2, b2, M0c2, shc2, g2, be2, sbn2, z_mid,
        nullptr, in8x, M0a1, sha1, M0a2, sha2, zadd, (float*)d_out);
}
